// MT_GAT_topk_shareEn_multiple8_70712341561395
// MI455X (gfx1250) — matrix operand order repaired and checked by reading, not compiled, not run on hardware
//
#include <hip/hip_runtime.h>
#include <hip/hip_bf16.h>

// ---------------------------------------------------------------------------
// Problem constants (match reference)
// ---------------------------------------------------------------------------
#define NB      256          // graphs
#define NODES   264          // nodes per graph
#define NN      (NB * NODES) // 67584 nodes
#define FIN     263
#define HCH     64           // hidden channels
#define KSEL    132          // top-k per graph
#define FINPAD  288          // 263 padded to multiple of 32
#define KFLAT   (KSEL * HCH) // 8448
#define NEG_SLOPE 0.2f

typedef __attribute__((ext_vector_type(16))) _Float16 v16h;
typedef __attribute__((ext_vector_type(8)))  _Float16 v8h;
typedef __attribute__((ext_vector_type(8)))  float    v8f;

// ---------------------------------------------------------------------------
// WMMA fragment loaders (16x16x32 f16, this code's contiguous k order, wave32)
// A: row-major [M,K].  Lane l (m = l&15):
//   elems 0..15 = A[m][k0 + 16*(l>>4) + 0..15]
// B: pre-transposed Bt[n][k] (= B[k][n]).  Lane l (n = l&15):
//   elems 0..15 = Bt[n][k0 + 16*(l>>4) + 0..15]   (one contiguous 32B load)
// ---------------------------------------------------------------------------
__device__ inline v16h load_a_frag(const _Float16* A, int lda, int m0, int k0, int lane) {
    int m  = m0 + (lane & 15);
    int kb = k0 + ((lane >> 4) << 4);
    const _Float16* p = A + (size_t)m * lda + kb;
    v8h lo = *(const v8h*)p;
    v8h hi = *(const v8h*)(p + 8);
    v16h a;
    #pragma unroll
    for (int j = 0; j < 8; ++j) { a[j] = lo[j]; a[8 + j] = hi[j]; }
    return a;
}

__device__ inline v16h load_b_frag(const _Float16* Bt, int ldb, int n0, int k0, int lane) {
    int n  = n0 + (lane & 15);
    int kh = k0 + ((lane >> 4) << 4);
    return *(const v16h*)(Bt + (size_t)n * ldb + kh);
}

// ---------------------------------------------------------------------------
// f16 WMMA GEMM:  C[M,64] = A[M,K] * Bt[64,K]^T  (+bias, +relu)
// One wave owns a full 16x64 row-tile: A fragment loaded once per k-step,
// reused across 4 back-to-back WMMAs (4 accumulators). block = 128 (4 waves,
// 4 consecutive M-tiles per block).
// ---------------------------------------------------------------------------
__global__ void gemm_wmma_f16(const _Float16* __restrict__ A,
                              const _Float16* __restrict__ Bt,
                              float* __restrict__ C,
                              int M, int K, int lda, int ldb, int ldc,
                              const float* __restrict__ bias, int do_relu) {
    int wave = threadIdx.x >> 5;
    int lane = threadIdx.x & 31;
    int m0 = (blockIdx.x * 4 + wave) * 16;
    if (m0 >= M) return;

    v8f acc0 = {}, acc1 = {}, acc2 = {}, acc3 = {};
    for (int k0 = 0; k0 < K; k0 += 32) {
        if (k0 + 64 < K)  // warm L2 ahead of time (global_prefetch_b8)
            __builtin_prefetch(A + (size_t)(m0 + (lane & 15)) * lda + k0 + 64, 0, 3);
        v16h a  = load_a_frag(A, lda, m0, k0, lane);
        v16h b0 = load_b_frag(Bt, ldb,  0, k0, lane);
        v16h b1 = load_b_frag(Bt, ldb, 16, k0, lane);
        v16h b2 = load_b_frag(Bt, ldb, 32, k0, lane);
        v16h b3 = load_b_frag(Bt, ldb, 48, k0, lane);
        acc0 = __builtin_amdgcn_wmma_f32_16x16x32_f16(false, a, false, b0, (short)0, acc0, false, false);
        acc1 = __builtin_amdgcn_wmma_f32_16x16x32_f16(false, a, false, b1, (short)0, acc1, false, false);
        acc2 = __builtin_amdgcn_wmma_f32_16x16x32_f16(false, a, false, b2, (short)0, acc2, false, false);
        acc3 = __builtin_amdgcn_wmma_f32_16x16x32_f16(false, a, false, b3, (short)0, acc3, false, false);
    }

    int rbase = m0 + ((lane >> 4) << 3);
    int cbase = lane & 15;
    #pragma unroll
    for (int nt = 0; nt < 4; ++nt) {
        v8f acc = (nt == 0) ? acc0 : (nt == 1) ? acc1 : (nt == 2) ? acc2 : acc3;
        int col = nt * 16 + cbase;
        float bv = bias ? bias[col] : 0.0f;
        #pragma unroll
        for (int r = 0; r < 8; ++r) {
            float v = acc[r] + bv;
            if (do_relu) v = v > 0.0f ? v : 0.0f;
            C[(size_t)(rbase + r) * ldc + col] = v;
        }
    }
}

// ---------------------------------------------------------------------------
// Prep kernels: f32 -> f16 pad/convert, weight transpose to [N,Kpad] f16
// ---------------------------------------------------------------------------
__global__ void pad_convert_f16(const float* __restrict__ X, _Float16* __restrict__ Y,
                                int rows, int cin, int cpad) {
    size_t i = (size_t)blockIdx.x * blockDim.x + threadIdx.x;
    if (i >= (size_t)rows * cpad) return;
    int r = (int)(i / cpad), c = (int)(i % cpad);
    Y[i] = (_Float16)((c < cin) ? X[(size_t)r * cin + c] : 0.0f);
}

__global__ void transpose_f16(const float* __restrict__ W, _Float16* __restrict__ Wt,
                              int Kin, int Nc, int Kpad) {
    size_t i = (size_t)blockIdx.x * blockDim.x + threadIdx.x;
    if (i >= (size_t)Nc * Kpad) return;
    int n = (int)(i / Kpad), k = (int)(i % Kpad);
    Wt[i] = (_Float16)((k < Kin) ? W[(size_t)k * Nc + n] : 0.0f);
}

// ---------------------------------------------------------------------------
// GAT attention pipeline (segment softmax over dst via atomics)
// ---------------------------------------------------------------------------
__device__ inline unsigned encf(float f) {
    unsigned u = __float_as_uint(f);
    return (u & 0x80000000u) ? ~u : (u | 0x80000000u);
}
__device__ inline float decf(unsigned u) {
    return (u & 0x80000000u) ? __uint_as_float(u & 0x7FFFFFFFu) : __uint_as_float(~u);
}
__device__ inline float lrelu(float v) { return v > 0.0f ? v : NEG_SLOPE * v; }

__global__ void node_dots_k(const float* __restrict__ h, const float* __restrict__ as,
                            const float* __restrict__ ad, float* __restrict__ asrc,
                            float* __restrict__ adst, int n) {
    int i = blockIdx.x * blockDim.x + threadIdx.x;
    if (i >= n) return;
    float sa = 0.f, sd = 0.f;
    #pragma unroll 8
    for (int c = 0; c < HCH; ++c) {
        float v = h[(size_t)i * HCH + c];
        sa += v * as[c]; sd += v * ad[c];
    }
    asrc[i] = sa; adst[i] = sd;
}

__global__ void init_conv_k(unsigned* __restrict__ menc, float* __restrict__ ssum,
                            float* __restrict__ agg, int n) {
    size_t i = (size_t)blockIdx.x * blockDim.x + threadIdx.x;
    if (i >= (size_t)n * HCH) return;
    agg[i] = 0.0f;
    if (i < (size_t)n) { menc[i] = encf(-1e30f); ssum[i] = 0.0f; }
}

__global__ void edge_max_k(const int* __restrict__ src, const int* __restrict__ dst,
                           const float* __restrict__ asrc, const float* __restrict__ adst,
                           unsigned* __restrict__ menc, int E, int n) {
    int e = blockIdx.x * blockDim.x + threadIdx.x;
    if (e >= E + n) return;
    int s = (e < E) ? src[e] : e - E;
    int d = (e < E) ? dst[e] : e - E;
    atomicMax(&menc[d], encf(lrelu(asrc[s] + adst[d])));
}

__global__ void edge_sum_k(const int* __restrict__ src, const int* __restrict__ dst,
                           const float* __restrict__ asrc, const float* __restrict__ adst,
                           const unsigned* __restrict__ menc, float* __restrict__ ssum,
                           int E, int n) {
    int e = blockIdx.x * blockDim.x + threadIdx.x;
    if (e >= E + n) return;
    int s = (e < E) ? src[e] : e - E;
    int d = (e < E) ? dst[e] : e - E;
    float al = lrelu(asrc[s] + adst[d]);
    atomicAdd(&ssum[d], __expf(al - decf(menc[d])));
}

// warp per edge: 32 lanes x 2 channels
__global__ void edge_agg_k(const int* __restrict__ src, const int* __restrict__ dst,
                           const float* __restrict__ asrc, const float* __restrict__ adst,
                           const unsigned* __restrict__ menc, const float* __restrict__ ssum,
                           const float* __restrict__ h, float* __restrict__ agg,
                           int E, int n) {
    int w    = (int)(((size_t)blockIdx.x * blockDim.x + threadIdx.x) >> 5);
    int lane = threadIdx.x & 31;
    if (w >= E + n) return;
    int s = (w < E) ? src[w] : w - E;
    int d = (w < E) ? dst[w] : w - E;
    float al  = lrelu(asrc[s] + adst[d]);
    float att = __expf(al - decf(menc[d])) / (ssum[d] + 1e-16f);
    float2 hv = ((const float2*)(h + (size_t)s * HCH))[lane];
    atomicAdd(&agg[(size_t)d * HCH + 2 * lane],     att * hv.x);
    atomicAdd(&agg[(size_t)d * HCH + 2 * lane + 1], att * hv.y);
}

__global__ void finalize_k(const float* __restrict__ agg, const float* __restrict__ bias,
                           float* __restrict__ of32, _Float16* __restrict__ of16, int n) {
    size_t i = (size_t)blockIdx.x * blockDim.x + threadIdx.x;
    if (i >= (size_t)n * HCH) return;
    float v = agg[i] + bias[i & 63];
    v = v > 0.0f ? v : 0.0f;
    if (of32) of32[i] = v;
    if (of16) of16[i] = (_Float16)v;
}

// ---------------------------------------------------------------------------
// TopK heads
// ---------------------------------------------------------------------------
__global__ void pnorm_k(const float* __restrict__ pw, float* __restrict__ pn) {
    int p = threadIdx.x;
    if (p >= 8) return;
    float s = 0.f;
    for (int c = 0; c < HCH; ++c) { float v = pw[p * HCH + c]; s += v * v; }
    pn[p] = sqrtf(s);
}

__global__ void scores_k(const float* __restrict__ x11, const float* __restrict__ x22,
                         const float* __restrict__ pw, const float* __restrict__ pn,
                         float* __restrict__ sc, int n) {
    size_t i = (size_t)blockIdx.x * blockDim.x + threadIdx.x;
    if (i >= (size_t)8 * n) return;
    int p = (int)(i / n), node = (int)(i % n);
    const float* xh = (p < 4) ? x11 : x22;
    float s = 0.f;
    #pragma unroll 8
    for (int c = 0; c < HCH; ++c) s += xh[(size_t)node * HCH + c] * pw[p * HCH + c];
    s /= pn[p];
    sc[i] = 1.0f / (1.0f + __expf(-s));
}

// one block (256 threads) per graph: bitonic sort 512 entries in LDS, keep 132
__global__ void topk_k(const float* __restrict__ sc, float* __restrict__ ssel,
                       int* __restrict__ idxsel, float* __restrict__ out_perm,
                       float* __restrict__ out_sscore) {
    __shared__ float s[512];
    __shared__ int   id[512];
    int g = blockIdx.x;
    for (int i = threadIdx.x; i < 512; i += 256) {
        if (i < NODES) { s[i] = sc[(size_t)g * NODES + i]; id[i] = i; }
        else           { s[i] = -3.4e38f; id[i] = 1 << 20; }
    }
    __syncthreads();
    for (int k = 2; k <= 512; k <<= 1) {
        for (int j = k >> 1; j > 0; j >>= 1) {
            for (int i = threadIdx.x; i < 512; i += 256) {
                int ixj = i ^ j;
                if (ixj > i) {
                    bool desc = ((i & k) == 0);
                    float si = s[i], sj = s[ixj];
                    int   ii = id[i], ij = id[ixj];
                    bool sw = desc ? (si < sj || (si == sj && ii > ij))
                                   : (si > sj || (si == sj && ii < ij));
                    if (sw) { s[i] = sj; s[ixj] = si; id[i] = ij; id[ixj] = ii; }
                }
            }
            __syncthreads();
        }
    }
    for (int t = threadIdx.x; t < KSEL; t += 256) {
        int li = id[t]; float sv = s[t];
        ssel[g * KSEL + t]       = sv;
        idxsel[g * KSEL + t]     = li;
        out_perm[g * KSEL + t]   = (float)(g * NODES + li);
        out_sscore[g * KSEL + t] = 1.0f / (1.0f + __expf(-sv));
    }
}

__global__ void gather_k(const float* __restrict__ xh, const float* __restrict__ ssel,
                         const int* __restrict__ idxsel, _Float16* __restrict__ flat) {
    size_t i = (size_t)blockIdx.x * blockDim.x + threadIdx.x;
    if (i >= (size_t)NB * KSEL * HCH) return;
    int c  = (int)(i & 63);
    size_t gk = i >> 6;
    int k = (int)(gk % KSEL), g = (int)(gk / KSEL);
    int li = idxsel[g * KSEL + k];
    float v = xh[((size_t)g * NODES + li) * HCH + c] * ssel[g * KSEL + k];
    v = v > 0.0f ? v : 0.0f;   // relu(flat)
    flat[(size_t)g * KFLAT + k * HCH + c] = (_Float16)v;
}

__global__ void head_out_k(const float* __restrict__ h1, const float* __restrict__ lbw,
                           const float* __restrict__ lbb, float* __restrict__ out) {
    int b = blockIdx.x * blockDim.x + threadIdx.x;
    if (b >= NB) return;
    float s = 0.f;
    #pragma unroll 8
    for (int c = 0; c < HCH; ++c) {
        float v = h1[(size_t)b * HCH + c];
        v = v > 0.0f ? v : 0.0f;   // relu(h1)
        s += v * lbw[c];
    }
    out[b] = s + lbb[0];
}

// ---------------------------------------------------------------------------
// Host orchestration
// ---------------------------------------------------------------------------
extern "C" void kernel_launch(void* const* d_in, const int* in_sizes, int n_in,
                              void* d_out, int out_size, void* d_ws, size_t ws_size,
                              hipStream_t stream) {
    const float* x    = (const float*)d_in[0];
    const int*   ei   = (const int*)d_in[1];
    const int    E    = in_sizes[1] / 2;
    const int*   src  = ei;
    const int*   dst  = ei + E;
    const float* W0   = (const float*)d_in[4];
    const float* as0  = (const float*)d_in[5];
    const float* ad0  = (const float*)d_in[6];
    const float* b0   = (const float*)d_in[7];
    const float* W1   = (const float*)d_in[8];
    const float* as1  = (const float*)d_in[9];
    const float* ad1  = (const float*)d_in[10];
    const float* b1   = (const float*)d_in[11];
    const float* W2   = (const float*)d_in[12];
    const float* as2  = (const float*)d_in[13];
    const float* ad2  = (const float*)d_in[14];
    const float* b2   = (const float*)d_in[15];
    const float* pw   = (const float*)d_in[16];
    const float* law  = (const float*)d_in[17];
    const float* lab  = (const float*)d_in[18];
    const float* lbw  = (const float*)d_in[19];
    const float* lbb  = (const float*)d_in[20];
    float* out = (float*)d_out;

    // bump allocator over d_ws
    char* cur = (char*)d_ws;
    auto alloc = [&](size_t bytes) -> void* {
        void* p = cur;
        cur += (bytes + 255) & ~(size_t)255;
        return p;
    };
    _Float16* xpad  = (_Float16*)alloc((size_t)NN * FINPAD * 2);
    _Float16* w0t   = (_Float16*)alloc((size_t)HCH * FINPAD * 2);
    _Float16* w1t   = (_Float16*)alloc((size_t)HCH * HCH * 2);
    _Float16* w2t   = (_Float16*)alloc((size_t)HCH * HCH * 2);
    _Float16* lawT  = (_Float16*)alloc((size_t)8 * HCH * KFLAT * 2);
    float*    h     = (float*)alloc((size_t)NN * HCH * 4);
    float*    agg   = (float*)alloc((size_t)NN * HCH * 4);
    _Float16* h0f16 = (_Float16*)alloc((size_t)NN * HCH * 2);
    float*    x11   = (float*)alloc((size_t)NN * HCH * 4);
    float*    x22   = (float*)alloc((size_t)NN * HCH * 4);
    float*    asrc  = (float*)alloc((size_t)NN * 4);
    float*    adstb = (float*)alloc((size_t)NN * 4);
    unsigned* menc  = (unsigned*)alloc((size_t)NN * 4);
    float*    ssum  = (float*)alloc((size_t)NN * 4);
    float*    sc    = (float*)alloc((size_t)8 * NN * 4);
    float*    pn    = (float*)alloc(64);
    float*    ssel  = (float*)alloc((size_t)NB * KSEL * 4);
    int*      idxs  = (int*)alloc((size_t)NB * KSEL * 4);
    _Float16* flat  = (_Float16*)alloc((size_t)NB * KFLAT * 2);
    float*    h1    = (float*)alloc((size_t)NB * HCH * 4);
    (void)ws_size; (void)n_in; (void)out_size;

    const int Etot = E + NN;

    // ---- prep: f16 conversions / weight transposes ----
    {
        size_t tot = (size_t)NN * FINPAD;
        pad_convert_f16<<<(int)((tot + 255) / 256), 256, 0, stream>>>(x, xpad, NN, FIN, FINPAD);
        transpose_f16<<<(HCH * FINPAD + 255) / 256, 256, 0, stream>>>(W0, w0t, FIN, HCH, FINPAD);
        transpose_f16<<<(HCH * HCH + 255) / 256, 256, 0, stream>>>(W1, w1t, HCH, HCH, HCH);
        transpose_f16<<<(HCH * HCH + 255) / 256, 256, 0, stream>>>(W2, w2t, HCH, HCH, HCH);
        for (int p = 0; p < 8; ++p)
            transpose_f16<<<(HCH * KFLAT + 255) / 256, 256, 0, stream>>>(
                law + (size_t)p * KFLAT * HCH, lawT + (size_t)p * HCH * KFLAT, KFLAT, HCH, KFLAT);
        pnorm_k<<<1, 8, 0, stream>>>(pw, pn);
    }

    // ---- one GATConv layer ----
    auto run_conv = [&](const _Float16* Af16, int K, const _Float16* Wt,
                        const float* as, const float* ad, const float* bias,
                        float* of32, _Float16* of16) {
        gemm_wmma_f16<<<NN / 64, 128, 0, stream>>>(Af16, Wt, h, NN, K, K, K, HCH, nullptr, 0);
        node_dots_k<<<(NN + 255) / 256, 256, 0, stream>>>(h, as, ad, asrc, adstb, NN);
        init_conv_k<<<(NN * HCH + 255) / 256, 256, 0, stream>>>(menc, ssum, agg, NN);
        edge_max_k<<<(Etot + 255) / 256, 256, 0, stream>>>(src, dst, asrc, adstb, menc, E, NN);
        edge_sum_k<<<(Etot + 255) / 256, 256, 0, stream>>>(src, dst, asrc, adstb, menc, ssum, E, NN);
        edge_agg_k<<<(Etot + 7) / 8, 256, 0, stream>>>(src, dst, asrc, adstb, menc, ssum, h, agg, E, NN);
        finalize_k<<<(NN * HCH + 255) / 256, 256, 0, stream>>>(agg, bias, of32, of16, NN);
    };

    run_conv(xpad,  FINPAD, w0t, as0, ad0, b0, nullptr, h0f16);   // h0 (f16)
    run_conv(h0f16, HCH,    w1t, as1, ad1, b1, x11, nullptr);     // x11 (f32)
    run_conv(h0f16, HCH,    w2t, as2, ad2, b2, x22, nullptr);     // x22 (f32)

    // ---- 8 TopK heads ----
    scores_k<<<(8 * NN + 255) / 256, 256, 0, stream>>>(x11, x22, pw, pn, sc, NN);
    const int HEAD_OUT = NB + 2 * NB * KSEL;   // 256 + 33792 + 33792
    for (int p = 0; p < 8; ++p) {
        const float* xh = (p < 4) ? x11 : x22;
        float* ob = out + (size_t)p * HEAD_OUT;
        topk_k<<<NB, 256, 0, stream>>>(sc + (size_t)p * NN, ssel, idxs,
                                       ob + NB, ob + NB + NB * KSEL);
        gather_k<<<(NB * KSEL * HCH + 255) / 256, 256, 0, stream>>>(xh, ssel, idxs, flat);
        gemm_wmma_f16<<<(NB / 16 + 3) / 4, 128, 0, stream>>>(flat, lawT + (size_t)p * HCH * KFLAT, h1,
                                                             NB, KFLAT, KFLAT, KFLAT, HCH,
                                                             lab + p * HCH, 0);
        head_out_k<<<1, 256, 0, stream>>>(h1, lbw + p * HCH, lbb + p, ob);
    }
}
